// AdaptiveSpatialAttention_9663676416299
// MI455X (gfx1250) — matrix operand order repaired and checked by reading, not compiled, not run on hardware
//
#include <hip/hip_runtime.h>
#include <hip/hip_bf16.h>

typedef __attribute__((ext_vector_type(16))) _Float16 v16h;
typedef __attribute__((ext_vector_type(8)))  float    v8f;

#define BB 16
#define HH 64
#define WWD 64
#define CC 192
#define LL (HH*WWD)      /* 4096 */
#define C3 (3*CC)        /* 576  */
#define SPS 8
#define WS 64
#define ML (BB*LL)       /* 65536 */

// ---------------------------------------------------------------------------
// Kernel 1: relative-position MLP tables (2 branches x 225 x 3 heads) and the
// constant channel gate (the C-Map LayerNorm is over a size-1 axis, so it
// reduces to a data-independent per-channel constant).
// ---------------------------------------------------------------------------
__device__ __forceinline__ void ln_relu(float* v, int n, const float* g, const float* be, float eps) {
    float m = 0.f;
    for (int i = 0; i < n; ++i) m += v[i];
    m /= (float)n;
    float var = 0.f;
    for (int i = 0; i < n; ++i) { float d = v[i] - m; var += d * d; }
    var /= (float)n;
    float inv = rsqrtf(var + eps);
    for (int i = 0; i < n; ++i) {
        float y = (v[i] - m) * inv * g[i] + be[i];
        v[i] = fmaxf(y, 0.f);
    }
}

__global__ void posmlp_gate_kernel(const float* __restrict__ pw0, const float* __restrict__ pb0,
                                   const float* __restrict__ g1,  const float* __restrict__ be1,
                                   const float* __restrict__ pw1, const float* __restrict__ pb1,
                                   const float* __restrict__ g2,  const float* __restrict__ be2,
                                   const float* __restrict__ pw2, const float* __restrict__ pb2,
                                   const float* __restrict__ g3,  const float* __restrict__ be3,
                                   const float* __restrict__ pw3, const float* __restrict__ pb3,
                                   const float* __restrict__ ci_ln_b, const float* __restrict__ ci_w2,
                                   const float* __restrict__ ci_b2,
                                   float* __restrict__ table, float* __restrict__ gate)
{
    int tid = threadIdx.x;
    if (tid < 450) {
        int br = tid / 225;
        int t  = tid - br * 225;
        float r0 = (float)(t / 15) - 7.0f;
        float r1 = (float)(t % 15) - 7.0f;
        float v12[12];
        for (int j = 0; j < 12; ++j)
            v12[j] = r0 * pw0[br*24 + j] + r1 * pw0[br*24 + 12 + j] + pb0[br*12 + j];
        ln_relu(v12, 12, g1 + br*12, be1 + br*12, 1e-3f);
        float v6[6];
        for (int k = 0; k < 6; ++k) {
            float s = pb1[br*6 + k];
            for (int j = 0; j < 12; ++j) s += v12[j] * pw1[br*72 + j*6 + k];
            v6[k] = s;
        }
        ln_relu(v6, 6, g2 + br*6, be2 + br*6, 1e-3f);
        float v6b[6];
        for (int k = 0; k < 6; ++k) {
            float s = pb2[br*6 + k];
            for (int j = 0; j < 6; ++j) s += v6[j] * pw2[br*36 + j*6 + k];
            v6b[k] = s;
        }
        ln_relu(v6b, 6, g3 + br*6, be3 + br*6, 1e-3f);
        for (int h = 0; h < 3; ++h) {
            float s = pb3[br*3 + h];
            for (int j = 0; j < 6; ++j) s += v6b[j] * pw3[br*18 + j*3 + h];
            table[(br*225 + t)*3 + h] = s;
        }
    }
    if (tid < CC) {
        float r = fmaxf(ci_ln_b[0], 0.f);
        float s = ci_b2[tid];
        for (int k = 0; k < 24; ++k) s += r * ci_w2[k*CC + tid];
        gate[tid] = 1.f / (1.f + expf(-s));
    }
}

// ---------------------------------------------------------------------------
// Kernel 2: pre-pack a weight matrix B[K,N] (f32, row-major) into f16 WMMA
// B-fragments: fragment f = kt*(N/16) + nt holds 32 lanes x 16 halfs, where
// lane's 16 halfs are B[kt*32 + kbB + j, nt*16 + (lane&15)], kbB = hi16*16.
// Done once per launch; lets the GEMM load B as one aligned 32-byte v16h.
// ---------------------------------------------------------------------------
__global__ void pack_b_f16(const float* __restrict__ B, _Float16* __restrict__ out, int N, int K)
{
    int idx = blockIdx.x * blockDim.x + threadIdx.x;
    int total = (K >> 5) * (N >> 4) * 32;
    if (idx >= total) return;
    int lane = idx & 31;
    int f    = idx >> 5;
    int ntn  = N >> 4;
    int kt   = f / ntn;
    int nt   = f - kt * ntn;
    int kbB  = (lane >= 16) ? 16 : 0;
    int col  = nt * 16 + (lane & 15);
    _Float16* o = out + (size_t)idx * 16;
    #pragma unroll
    for (int j = 0; j < 16; ++j)
        o[j] = (_Float16)B[(size_t)(kt * 32 + kbB + j) * N + col];
}

// ---------------------------------------------------------------------------
// Kernel 3/7: WMMA f16 GEMM, C[M,N] = A[M,K] @ Bpacked (+ bias[N]).
// One wave computes a register-blocked 16x64 tile (4 accumulators). Per k-step:
// one clause of loads (A as 4 b128, all four B fragments as 8 b128 at fixed
// 1 KB immediate offsets), then 4 back-to-back independent v_wmma issues.
// Requires M%16==0, N%64==0, K%32==0; grid sized so every wave has a tile.
// ---------------------------------------------------------------------------
__global__ void wmma_gemm_f16(const float* __restrict__ A, const _Float16* __restrict__ Bp,
                              const float* __restrict__ bias, float* __restrict__ C,
                              int M, int N, int K)
{
    int wave = threadIdx.x >> 5;
    int lane = threadIdx.x & 31;
    int tile = blockIdx.x * (blockDim.x >> 5) + wave;
    int ntn  = N >> 6;                       // 64-wide column tiles
    int ntm  = M >> 4;
    if (tile >= ntm * ntn) return;           // wave-uniform, EXEC stays all-1s
    int ntn16 = N >> 4;
    int m0 = (tile / ntn) << 4;
    int nt0 = (tile % ntn) << 2;             // first 16-col fragment index
    int ln15 = lane & 15;
    int hi   = lane >> 4;
    int kbA  = hi * 16;      // A lanes 0-15: K 0..15; lanes 16-31: K 16..31
    int rowA = m0 + ln15;

    const float*    ap = A + (size_t)rowA * K + kbA;
    const _Float16* bp = Bp + ((size_t)nt0 * 32 + lane) * 16;    // this lane's frag nt0
    size_t bstride = (size_t)ntn16 * 512;                        // halfs per k-tile row

    v8f z = {0.f,0.f,0.f,0.f,0.f,0.f,0.f,0.f};
    v8f acc0 = z, acc1 = z, acc2 = z, acc3 = z;
    for (int kk = 0; kk < K; kk += 32) {
        const float4* ap4 = (const float4*)ap;
        float4 a0 = ap4[0], a1 = ap4[1], a2 = ap4[2], a3 = ap4[3];
        v16h b0 = *(const v16h*)(bp);
        v16h b1 = *(const v16h*)(bp + 512);       // +1024 B
        v16h b2 = *(const v16h*)(bp + 1024);      // +2048 B
        v16h b3 = *(const v16h*)(bp + 1536);      // +3072 B
        v16h a;
        a[0]  = (_Float16)a0.x; a[1]  = (_Float16)a0.y; a[2]  = (_Float16)a0.z; a[3]  = (_Float16)a0.w;
        a[4]  = (_Float16)a1.x; a[5]  = (_Float16)a1.y; a[6]  = (_Float16)a1.z; a[7]  = (_Float16)a1.w;
        a[8]  = (_Float16)a2.x; a[9]  = (_Float16)a2.y; a[10] = (_Float16)a2.z; a[11] = (_Float16)a2.w;
        a[12] = (_Float16)a3.x; a[13] = (_Float16)a3.y; a[14] = (_Float16)a3.z; a[15] = (_Float16)a3.w;
        acc0 = __builtin_amdgcn_wmma_f32_16x16x32_f16(false, a, false, b0, (short)0, acc0, false, false);
        acc1 = __builtin_amdgcn_wmma_f32_16x16x32_f16(false, a, false, b1, (short)0, acc1, false, false);
        acc2 = __builtin_amdgcn_wmma_f32_16x16x32_f16(false, a, false, b2, (short)0, acc2, false, false);
        acc3 = __builtin_amdgcn_wmma_f32_16x16x32_f16(false, a, false, b3, (short)0, acc3, false, false);
        ap += 32;
        bp += bstride;
    }
    v8f accs[4] = {acc0, acc1, acc2, acc3};
    #pragma unroll
    for (int t = 0; t < 4; ++t) {
        int col = (nt0 + t) * 16 + ln15;
        float bz = bias ? bias[col] : 0.f;
        #pragma unroll
        for (int i = 0; i < 8; ++i) {
            int m = m0 + i + hi * 8;
            C[(size_t)m * N + col] = accs[t][i] + bz;
        }
    }
}

// ---------------------------------------------------------------------------
// Kernel 4: windowed attention. One wave per (branch, head, window).
// S = scale*Q@K^T + relpos-bias (WMMA), softmax in LDS (per-wave, in-order DS),
// O = P@V (WMMA). Head dim = 32 matches one 16x16x32 WMMA K-depth exactly.
// K and V fragments are hoisted into registers (they depend only on nt/kc).
// ---------------------------------------------------------------------------
__global__ void attn_kernel(const float* __restrict__ qkv, const float* __restrict__ table,
                            float* __restrict__ att)
{
    __shared__ float S[2][WS][WS + 1];
    int wave = threadIdx.x >> 5;
    int lane = threadIdx.x & 31;
    int gid  = blockIdx.x * 2 + wave;        // 0..6143
    int branch = gid / 3072;
    int rem    = gid % 3072;
    int head   = rem / 1024;
    int widx   = rem % 1024;
    int b   = widx >> 6;
    int win = widx & 63;
    int wh  = win >> 3, ww = win & 7;
    int cb  = branch * 96 + head * 32;       // channel base of this head
    int ln15 = lane & 15;
    int hi   = lane >> 4;
    int kbA  = hi * 16;
    int kbB  = hi * 16;
    float (*Sw)[WS + 1] = S[wave];
    const float scale = 0.17677669529663687f;    // 32^-0.5

    // token index (0..63) inside window -> row in [B, L]
    #define ROWOF(t) (b * LL + (wh * SPS + ((t) >> 3)) * WWD + (ww * SPS + ((t) & 7)))

    // ---- Phase 1: S = scale*Q @ K^T + bias -> LDS ----
    {
        v16h kf[4];
        #pragma unroll
        for (int nt = 0; nt < 4; ++nt) {
            int n = nt * 16 + ln15;
            const float4* kp4 = (const float4*)(qkv + (size_t)ROWOF(n) * C3 + CC + cb + kbB); // k slice
            float4 k0 = kp4[0], k1 = kp4[1], k2 = kp4[2], k3 = kp4[3];
            v16h bv;
            bv[0]  = (_Float16)k0.x; bv[1]  = (_Float16)k0.y; bv[2]  = (_Float16)k0.z; bv[3]  = (_Float16)k0.w;
            bv[4]  = (_Float16)k1.x; bv[5]  = (_Float16)k1.y; bv[6]  = (_Float16)k1.z; bv[7]  = (_Float16)k1.w;
            bv[8]  = (_Float16)k2.x; bv[9]  = (_Float16)k2.y; bv[10] = (_Float16)k2.z; bv[11] = (_Float16)k2.w;
            bv[12] = (_Float16)k3.x; bv[13] = (_Float16)k3.y; bv[14] = (_Float16)k3.z; bv[15] = (_Float16)k3.w;
            kf[nt] = bv;
        }
        for (int mt = 0; mt < 4; ++mt) {
            int m = mt * 16 + ln15;
            const float4* qp4 = (const float4*)(qkv + (size_t)ROWOF(m) * C3 + cb + kbA);   // q slice
            float4 q0 = qp4[0], q1 = qp4[1], q2 = qp4[2], q3 = qp4[3];
            v16h a;
            a[0]  = (_Float16)(q0.x * scale); a[1]  = (_Float16)(q0.y * scale);
            a[2]  = (_Float16)(q0.z * scale); a[3]  = (_Float16)(q0.w * scale);
            a[4]  = (_Float16)(q1.x * scale); a[5]  = (_Float16)(q1.y * scale);
            a[6]  = (_Float16)(q1.z * scale); a[7]  = (_Float16)(q1.w * scale);
            a[8]  = (_Float16)(q2.x * scale); a[9]  = (_Float16)(q2.y * scale);
            a[10] = (_Float16)(q2.z * scale); a[11] = (_Float16)(q2.w * scale);
            a[12] = (_Float16)(q3.x * scale); a[13] = (_Float16)(q3.y * scale);
            a[14] = (_Float16)(q3.z * scale); a[15] = (_Float16)(q3.w * scale);
            #pragma unroll
            for (int nt = 0; nt < 4; ++nt) {
                v8f acc = {0.f,0.f,0.f,0.f,0.f,0.f,0.f,0.f};
                acc = __builtin_amdgcn_wmma_f32_16x16x32_f16(false, a, false, kf[nt], (short)0, acc, false, false);
                #pragma unroll
                for (int i = 0; i < 8; ++i) {
                    int mm = mt * 16 + i + hi * 8;
                    int nn = nt * 16 + ln15;
                    int ridx = ((mm >> 3) - (nn >> 3) + 7) * 15 + ((mm & 7) - (nn & 7) + 7);
                    Sw[mm][nn] = acc[i] + table[(branch * 225 + ridx) * 3 + head];
                }
            }
        }
    }

    // ---- Phase 2: softmax rows (2 rows per lane; same-wave DS is in-order) ----
    #pragma unroll
    for (int rr = 0; rr < 2; ++rr) {
        int r = lane + rr * 32;
        float mx = -1e30f;
        for (int j = 0; j < WS; ++j) mx = fmaxf(mx, Sw[r][j]);
        float sum = 0.f;
        for (int j = 0; j < WS; ++j) { float e = expf(Sw[r][j] - mx); Sw[r][j] = e; sum += e; }
        float inv = 1.f / sum;
        for (int j = 0; j < WS; ++j) Sw[r][j] *= inv;
    }

    // ---- Phase 3: O = P @ V  (V fragments hoisted; depend only on kc,nt) ----
    {
        v16h vf[4];
        #pragma unroll
        for (int kc2 = 0; kc2 < 2; ++kc2) {
            #pragma unroll
            for (int nt = 0; nt < 2; ++nt) {
                int n = nt * 16 + ln15;                              // output dim 0..31
                v16h bv;
                #pragma unroll
                for (int j = 0; j < 16; ++j) {
                    int t = kc2 * 32 + kbB + j;
                    bv[j] = (_Float16)qkv[(size_t)ROWOF(t) * C3 + 2 * CC + cb + n];  // v slice
                }
                vf[kc2 * 2 + nt] = bv;
            }
        }
        for (int mt = 0; mt < 4; ++mt) {
            v8f z = {0.f,0.f,0.f,0.f,0.f,0.f,0.f,0.f};
            v8f acc[2] = {z, z};
            int m = mt * 16 + ln15;
            #pragma unroll
            for (int kc2 = 0; kc2 < 2; ++kc2) {
                v16h a;
                #pragma unroll
                for (int j = 0; j < 8; ++j) {
                    a[j]     = (_Float16)Sw[m][kc2 * 32 + kbA + j];
                    a[8 + j] = (_Float16)Sw[m][kc2 * 32 + kbA + 8 + j];
                }
                #pragma unroll
                for (int nt = 0; nt < 2; ++nt)
                    acc[nt] = __builtin_amdgcn_wmma_f32_16x16x32_f16(false, a, false, vf[kc2 * 2 + nt],
                                                                     (short)0, acc[nt], false, false);
            }
            #pragma unroll
            for (int nt = 0; nt < 2; ++nt) {
                int n = nt * 16 + ln15;
                #pragma unroll
                for (int i = 0; i < 8; ++i) {
                    int mm = mt * 16 + i + hi * 8;
                    att[(size_t)ROWOF(mm) * CC + cb + n] = acc[nt][i];
                }
            }
        }
    }
    #undef ROWOF
}

// ---------------------------------------------------------------------------
// Kernel 5: spatial interaction map. One block per (b, h) row of 64 pixels.
// LN over W (matches si_ln_g shape 64), then project to scalar per pixel.
// ---------------------------------------------------------------------------
__global__ void spatial_kernel(const float* __restrict__ att, const float* __restrict__ sw1,
                               const float* __restrict__ sb1, const float* __restrict__ lng,
                               const float* __restrict__ lnb, const float* __restrict__ sw2,
                               const float* __restrict__ sb2, float* __restrict__ smap)
{
    __shared__ float w1s[CC * 12];
    __shared__ float si[12][WWD];
    __shared__ float mn[12], iv[12];
    int w  = threadIdx.x;        // 64 threads
    int bh = blockIdx.x;         // b*64 + h  -> bh*64 + w == b*L + h*64 + w
    for (int i = w; i < CC * 12; i += WWD) w1s[i] = sw1[i];
    __syncthreads();

    float acc[12];
    #pragma unroll
    for (int d = 0; d < 12; ++d) acc[d] = sb1[d];
    const float* ap = att + (size_t)(bh * WWD + w) * CC;
    for (int c = 0; c < CC; ++c) {
        float av = ap[c];
        #pragma unroll
        for (int d = 0; d < 12; ++d) acc[d] += av * w1s[c * 12 + d];
    }
    #pragma unroll
    for (int d = 0; d < 12; ++d) si[d][w] = acc[d];
    __syncthreads();

    if (w < 12) {
        float m = 0.f;
        for (int j = 0; j < WWD; ++j) m += si[w][j];
        m *= (1.f / WWD);
        float v = 0.f;
        for (int j = 0; j < WWD; ++j) { float d = si[w][j] - m; v += d * d; }
        v *= (1.f / WWD);
        mn[w] = m;
        iv[w] = rsqrtf(v + 1e-5f);
    }
    __syncthreads();

    float g = lng[w], bb = lnb[w];
    float sm = sb2[0];
    #pragma unroll
    for (int d = 0; d < 12; ++d) {
        float t = (si[d][w] - mn[d]) * iv[d] * g + bb;
        sm += fmaxf(t, 0.f) * sw2[d];
    }
    smap[bh * WWD + w] = sm;
}

// ---------------------------------------------------------------------------
// Kernel 6: fused depthwise 3x3 conv (SAME) + BN + ReLU + dual gating + sum,
// written in place into the attened buffer (becomes the projection input).
// ---------------------------------------------------------------------------
__global__ void combine_dwconv_kernel(float* __restrict__ att, const float* __restrict__ qkv,
                                      const float* __restrict__ dwk, const float* __restrict__ dwb,
                                      const float* __restrict__ bng, const float* __restrict__ bnb,
                                      const float* __restrict__ gate, const float* __restrict__ smap)
{
    int idx = blockIdx.x * blockDim.x + threadIdx.x;
    if (idx >= BB * LL * CC) return;
    int c  = idx % CC;
    int bl = idx / CC;
    int l  = bl % LL;
    int b  = bl / LL;
    int h  = l >> 6, w = l & 63;
    float sum = dwb[c];
    #pragma unroll
    for (int ky = 0; ky < 3; ++ky) {
        int hh = h + ky - 1;
        if (hh < 0 || hh >= HH) continue;
        #pragma unroll
        for (int kx = 0; kx < 3; ++kx) {
            int wwp = w + kx - 1;
            if (wwp < 0 || wwp >= WWD) continue;
            sum += qkv[(size_t)(b * LL + hh * WWD + wwp) * C3 + 2 * CC + c] * dwk[(ky * 3 + kx) * CC + c];
        }
    }
    float conv = fmaxf(sum * (bng[c] * rsqrtf(1.0f + 1e-3f)) + bnb[c], 0.f);
    float sg   = 1.f / (1.f + expf(-smap[bl]));
    att[idx]   = att[idx] * gate[c] + conv * sg;
}

// ---------------------------------------------------------------------------
extern "C" void kernel_launch(void* const* d_in, const int* in_sizes, int n_in,
                              void* d_out, int out_size, void* d_ws, size_t ws_size,
                              hipStream_t stream) {
    (void)in_sizes; (void)n_in; (void)out_size; (void)ws_size;
    const float* x      = (const float*)d_in[0];
    const float* w_qkv  = (const float*)d_in[1];
    const float* pw0 = (const float*)d_in[2];  const float* pb0 = (const float*)d_in[3];
    const float* g1  = (const float*)d_in[4];  const float* be1 = (const float*)d_in[5];
    const float* pw1 = (const float*)d_in[6];  const float* pb1 = (const float*)d_in[7];
    const float* g2  = (const float*)d_in[8];  const float* be2 = (const float*)d_in[9];
    const float* pw2 = (const float*)d_in[10]; const float* pb2 = (const float*)d_in[11];
    const float* g3  = (const float*)d_in[12]; const float* be3 = (const float*)d_in[13];
    const float* pw3 = (const float*)d_in[14]; const float* pb3 = (const float*)d_in[15];
    const float* dwk = (const float*)d_in[16]; const float* dwb = (const float*)d_in[17];
    const float* bng = (const float*)d_in[18]; const float* bnb = (const float*)d_in[19];
    /* ci_w1 (20), ci_b1 (21), ci_ln_g (22) algebraically drop out (LN over size-1 axis) */
    const float* ci_ln_b = (const float*)d_in[23];
    const float* ci_w2   = (const float*)d_in[24];
    const float* ci_b2   = (const float*)d_in[25];
    const float* si_w1   = (const float*)d_in[26]; const float* si_b1 = (const float*)d_in[27];
    const float* si_ln_g = (const float*)d_in[28]; const float* si_ln_b = (const float*)d_in[29];
    const float* si_w2   = (const float*)d_in[30]; const float* si_b2   = (const float*)d_in[31];
    const float* w_proj  = (const float*)d_in[32]; const float* b_proj  = (const float*)d_in[33];
    float* out = (float*)d_out;

    float* ws    = (float*)d_ws;
    float*    table = ws;                       // 2*225*3 = 1350 floats
    float*    gate  = ws + 2048;                // 192 floats
    float*    smap  = ws + 4096;                // 65536 floats
    _Float16* packq = (_Float16*)(ws + 69632);  // 192*576 halfs (55296 floats)
    _Float16* packp = (_Float16*)(ws + 124928); // 192*192 halfs (18432 floats)
    float*    qkvb  = ws + 262144;              // 16*4096*576 = 37748736 floats
    float*    attb  = qkvb + (size_t)ML * C3;   // 16*4096*192 = 12582912 floats

    // 1. pos-MLP tables + constant channel gate
    posmlp_gate_kernel<<<1, 512, 0, stream>>>(pw0, pb0, g1, be1, pw1, pb1, g2, be2,
                                              pw2, pb2, g3, be3, pw3, pb3,
                                              ci_ln_b, ci_w2, ci_b2, table, gate);
    // 2. pre-pack both weight matrices into f16 WMMA fragments
    pack_b_f16<<<27, 256, 0, stream>>>(w_qkv, packq, C3, CC);    // (192/32)*(576/16)*32 = 6912
    pack_b_f16<<<9, 256, 0, stream>>>(w_proj, packp, CC, CC);    // (192/32)*(192/16)*32 = 2304
    // 3. qkv = x @ w_qkv   (65536 x 576 x 192): 4096*9 = 36864 wave-tiles / 8 per block
    wmma_gemm_f16<<<4608, 256, 0, stream>>>(x, packq, nullptr, qkvb, ML, C3, CC);
    // 4. windowed attention (6144 wave-problems, 2 waves per block)
    attn_kernel<<<3072, 64, 0, stream>>>(qkvb, table, attb);
    // 5. spatial map (one block per (b,h)) — uses pre-gating attened
    spatial_kernel<<<BB * HH, WWD, 0, stream>>>(attb, si_w1, si_b1, si_ln_g, si_ln_b, si_w2, si_b2, smap);
    // 6. fused dwconv + BN + relu + gating + sum (in place into attb)
    combine_dwconv_kernel<<<(BB * LL * CC + 255) / 256, 256, 0, stream>>>(attb, qkvb, dwk, dwb,
                                                                         bng, bnb, gate, smap);
    // 7. out = combined @ w_proj + b_proj  (65536 x 192 x 192): 4096*3 = 12288 tiles
    wmma_gemm_f16<<<1536, 256, 0, stream>>>(attb, packp, b_proj, out, ML, CC, CC);
}
